// SimpleAttention_v2_70342974374217
// MI455X (gfx1250) — matrix operand order repaired and checked by reading, not compiled, not run on hardware
//
#include <hip/hip_runtime.h>
#include <hip/hip_bf16.h>
#include <math.h>

#define BATCH 4
#define SEQ   2048
#define DIM   1024
#define SCALE 0.03125f   // 1/sqrt(1024)

typedef __attribute__((ext_vector_type(16))) __bf16 v16bf;
typedef __attribute__((ext_vector_type(8)))  float  v8f;

// ---- CDNA5 WMMA fragment loaders (wave32, 16x16x32 bf16) -------------------
// A (16xK=32, row-major source): lane l -> row m=l&15, half h=l>>4.
//   elements 0..7  : K = k0 + 8*h + e ; elements 8..15 : K = k0 + 16 + 8*h + (e-8)
__device__ __forceinline__ v16bf load_a16(const __bf16* base, int ld, int row0,
                                          int k0, int lane) {
  const int m = lane & 15, h = lane >> 4;
  const __bf16* p = base + (size_t)(row0 + m) * ld + k0 + 8 * h;
  v16bf a;
  ((uint4*)&a)[0] = *(const uint4*)(p);
  ((uint4*)&a)[1] = *(const uint4*)(p + 16);
  return a;
}
// B (K=32 x 16): lane l -> col n=l&15, half h=l>>4; elements 0..7: K = k0+8h+e; 8..15: K = k0+16+8h+(e-8).
// Source laid out with the n-row contiguous in K (i.e. B^T row-major).
__device__ __forceinline__ v16bf load_b16(const __bf16* baseT, int ld, int col0,
                                          int k0, int lane) {
  const int n = lane & 15, h = lane >> 4;
  const __bf16* p = baseT + (size_t)(col0 + n) * ld + k0 + 8 * h;
  v16bf b;
  ((uint4*)&b)[0] = *(const uint4*)(p);
  ((uint4*)&b)[1] = *(const uint4*)(p + 16);
  return b;
}
// C/D: VGPR r, lane l -> m = r + 8*(l>>4), n = l&15.

// ---- fp32 -> bf16 convert ---------------------------------------------------
__global__ void cvt_f32_bf16(const float* __restrict__ in, __bf16* __restrict__ out, int n) {
  int i = blockIdx.x * 256 + threadIdx.x;
  if (i < n) out[i] = (__bf16)in[i];
}

// ---- QKV projection: each wave computes a 16x64 strip (4 N-tiles, shared A) -
__global__ void qkv_gemm(const __bf16* __restrict__ xh, const __bf16* __restrict__ Wh,
                         __bf16* __restrict__ Qh, __bf16* __restrict__ Kh,
                         __bf16* __restrict__ Vth) {
  const int lane = threadIdx.x & 31;
  const int wv   = threadIdx.x >> 5;                 // 4 waves / block
  const int n0   = (blockIdx.x * 4 + wv) * 64;       // 64-wide output strip
  const int m0   = blockIdx.y * 16;                  // token tile (b*S+s flat)
  const int z    = blockIdx.z;                       // 0:Q 1:K 2:V
  const __bf16* W = Wh + (size_t)z * DIM * DIM;      // W[o][d], contiguous in d=K
  v8f acc0 = {}, acc1 = {}, acc2 = {}, acc3 = {};
  for (int k0 = 0; k0 < DIM; k0 += 32) {
    v16bf a = load_a16(xh, DIM, m0, k0, lane);       // shared across 4 tiles
    v16bf b0 = load_b16(W, DIM, n0,      k0, lane);
    v16bf b1 = load_b16(W, DIM, n0 + 16, k0, lane);
    v16bf b2 = load_b16(W, DIM, n0 + 32, k0, lane);
    v16bf b3 = load_b16(W, DIM, n0 + 48, k0, lane);
    acc0 = __builtin_amdgcn_wmma_f32_16x16x32_bf16(false, a, false, b0, (short)0, acc0, false, false);
    acc1 = __builtin_amdgcn_wmma_f32_16x16x32_bf16(false, a, false, b1, (short)0, acc1, false, false);
    acc2 = __builtin_amdgcn_wmma_f32_16x16x32_bf16(false, a, false, b2, (short)0, acc2, false, false);
    acc3 = __builtin_amdgcn_wmma_f32_16x16x32_bf16(false, a, false, b3, (short)0, acc3, false, false);
  }
  const int h = lane >> 4, n = lane & 15;
  v8f accs[4] = {acc0, acc1, acc2, acc3};
  if (z == 0) {
    for (int j = 0; j < 4; ++j)
      for (int r = 0; r < 8; ++r)
        Qh[(size_t)(m0 + r + 8 * h) * DIM + n0 + 16 * j + n] = (__bf16)accs[j][r];
  } else if (z == 1) {
    for (int j = 0; j < 4; ++j)
      for (int r = 0; r < 8; ++r)
        Kh[(size_t)(m0 + r + 8 * h) * DIM + n0 + 16 * j + n] = (__bf16)accs[j][r];
  } else {
    const int bidx = m0 / SEQ;                       // 16-row tile never crosses batch
    const int s0   = m0 - bidx * SEQ;
    __bf16* Vb = Vth + (size_t)bidx * DIM * SEQ;     // Vt[b][d][s]
    for (int j = 0; j < 4; ++j)
      for (int r = 0; r < 8; ++r)
        Vb[(size_t)(n0 + 16 * j + n) * SEQ + s0 + r + 8 * h] = (__bf16)accs[j][r];
  }
}

// ---- attention: one block = (batch b, 16 query rows); 8 waves, 192KB LDS ---
__global__ void attention(const __bf16* __restrict__ Qh, const __bf16* __restrict__ Kh,
                          const __bf16* __restrict__ Vth, float* __restrict__ out) {
  extern __shared__ __align__(16) char smem[];
  float*  s_lds = (float*)smem;                      // [16][SEQ] scores/weights
  __bf16* p_lds = (__bf16*)(smem + 16 * SEQ * 4);    // [16][SEQ] final probs
  const int lane = threadIdx.x & 31;
  const int wv   = threadIdx.x >> 5;                 // 0..7
  const int q0   = blockIdx.x * 16;
  const int b    = blockIdx.y;
  const __bf16* Qb = Qh + (size_t)b * SEQ * DIM;
  const __bf16* Kb = Kh + (size_t)b * SEQ * DIM;
  const __bf16* Vb = Vth + (size_t)b * DIM * SEQ;
  const int h = lane >> 4, n = lane & 15;

  // Phase 1: S[16][SEQ] = Q_tile @ K^T. 32 groups of 4 key tiles over 8 waves.
  for (int g = wv; g < SEQ / 64; g += 8) {
    const int kt0 = g * 64;                          // 4 key tiles: kt0 .. kt0+48
    v8f acc0 = {}, acc1 = {}, acc2 = {}, acc3 = {};
    for (int d0 = 0; d0 < DIM; d0 += 32) {
      v16bf a  = load_a16(Qb, DIM, q0, d0, lane);    // shared A (Q rows)
      v16bf b0 = load_b16(Kb, DIM, kt0,      d0, lane);   // B(d,k)=K[k][d]
      v16bf b1 = load_b16(Kb, DIM, kt0 + 16, d0, lane);
      v16bf b2 = load_b16(Kb, DIM, kt0 + 32, d0, lane);
      v16bf b3 = load_b16(Kb, DIM, kt0 + 48, d0, lane);
      acc0 = __builtin_amdgcn_wmma_f32_16x16x32_bf16(false, a, false, b0, (short)0, acc0, false, false);
      acc1 = __builtin_amdgcn_wmma_f32_16x16x32_bf16(false, a, false, b1, (short)0, acc1, false, false);
      acc2 = __builtin_amdgcn_wmma_f32_16x16x32_bf16(false, a, false, b2, (short)0, acc2, false, false);
      acc3 = __builtin_amdgcn_wmma_f32_16x16x32_bf16(false, a, false, b3, (short)0, acc3, false, false);
    }
    v8f accs[4] = {acc0, acc1, acc2, acc3};
    for (int j = 0; j < 4; ++j)
      for (int r = 0; r < 8; ++r)
        s_lds[(size_t)(r + 8 * h) * SEQ + kt0 + 16 * j + n] = accs[j][r];
  }
  __syncthreads();

  // Phase 2: double softmax (reference semantics). Each wave owns 2 rows.
  for (int rr = 0; rr < 2; ++rr) {
    const int row = wv * 2 + rr;
    float* srow = s_lds + (size_t)row * SEQ;
    const int q = q0 + row;
    // softmax #1 over ALL keys: w = softmax(s * 1/sqrt(d))
    float mx = -3.0e38f;
    for (int c = lane; c < SEQ; c += 32) mx = fmaxf(mx, srow[c]);
    for (int o = 16; o > 0; o >>= 1) mx = fmaxf(mx, __shfl_xor(mx, o));
    float z1 = 0.f;
    for (int c = lane; c < SEQ; c += 32) {
      float e = __expf((srow[c] - mx) * SCALE);
      srow[c] = e; z1 += e;
    }
    for (int o = 16; o > 0; o >>= 1) z1 += __shfl_xor(z1, o);
    const float iz1 = 1.0f / z1;
    // mask WEIGHTS (k>q -> -inf => exp=0); softmax #2 of w/sqrt(d) over k<=q
    float z2 = 0.f;
    for (int c = lane; c < SEQ; c += 32) {
      float w  = srow[c] * iz1;                      // in [0,1]: no max needed
      float pu = (c <= q) ? __expf(w * SCALE) : 0.f;
      srow[c] = pu; z2 += pu;
    }
    for (int o = 16; o > 0; o >>= 1) z2 += __shfl_xor(z2, o);
    const float iz2 = 1.0f / z2;
    __bf16* prow = p_lds + (size_t)row * SEQ;
    for (int c = lane; c < SEQ; c += 32) prow[c] = (__bf16)(srow[c] * iz2);
  }
  __syncthreads();

  // Phase 3: O = P @ V. 16 groups of 4 output tiles over 8 waves.
  for (int g = wv; g < DIM / 64; g += 8) {
    const int n0 = g * 64;
    v8f acc0 = {}, acc1 = {}, acc2 = {}, acc3 = {};
    for (int k0 = 0; k0 < SEQ; k0 += 32) {
      v16bf a  = load_a16(p_lds, SEQ, 0, k0, lane);       // LDS A, shared x4
      v16bf b0 = load_b16(Vb, SEQ, n0,      k0, lane);    // B(k,n)=Vt[n][k]
      v16bf b1 = load_b16(Vb, SEQ, n0 + 16, k0, lane);
      v16bf b2 = load_b16(Vb, SEQ, n0 + 32, k0, lane);
      v16bf b3 = load_b16(Vb, SEQ, n0 + 48, k0, lane);
      acc0 = __builtin_amdgcn_wmma_f32_16x16x32_bf16(false, a, false, b0, (short)0, acc0, false, false);
      acc1 = __builtin_amdgcn_wmma_f32_16x16x32_bf16(false, a, false, b1, (short)0, acc1, false, false);
      acc2 = __builtin_amdgcn_wmma_f32_16x16x32_bf16(false, a, false, b2, (short)0, acc2, false, false);
      acc3 = __builtin_amdgcn_wmma_f32_16x16x32_bf16(false, a, false, b3, (short)0, acc3, false, false);
    }
    v8f accs[4] = {acc0, acc1, acc2, acc3};
    for (int j = 0; j < 4; ++j)
      for (int r = 0; r < 8; ++r)
        out[((size_t)b * SEQ + q0 + r + 8 * h) * DIM + n0 + 16 * j + n] = accs[j][r];
  }
}

extern "C" void kernel_launch(void* const* d_in, const int* in_sizes, int n_in,
                              void* d_out, int out_size, void* d_ws, size_t ws_size,
                              hipStream_t stream) {
  (void)in_sizes; (void)n_in; (void)out_size; (void)ws_size;
  const float* x  = (const float*)d_in[0];
  const float* Wq = (const float*)d_in[1];
  const float* Wk = (const float*)d_in[2];
  const float* Wv = (const float*)d_in[3];

  char* ws = (char*)d_ws;
  const size_t nx = (size_t)BATCH * SEQ * DIM;   // 8,388,608
  const size_t nw = (size_t)DIM * DIM;           // 1,048,576
  __bf16* xh  = (__bf16*)ws;                         // 16 MB
  __bf16* Wh  = (__bf16*)(ws + nx * 2);              // 6 MB (Wq,Wk,Wv)
  __bf16* Qh  = (__bf16*)(ws + nx * 2 + 3 * nw * 2); // 16 MB
  __bf16* Kh  = Qh + nx;                             // 16 MB
  __bf16* Vth = Kh + nx;                             // 16 MB (Vt[b][d][s])

  cvt_f32_bf16<<<(unsigned)((nx + 255) / 256), 256, 0, stream>>>(x, xh, (int)nx);
  cvt_f32_bf16<<<(unsigned)((nw + 255) / 256), 256, 0, stream>>>(Wq, Wh, (int)nw);
  cvt_f32_bf16<<<(unsigned)((nw + 255) / 256), 256, 0, stream>>>(Wk, Wh + nw, (int)nw);
  cvt_f32_bf16<<<(unsigned)((nw + 255) / 256), 256, 0, stream>>>(Wv, Wh + 2 * nw, (int)nw);

  dim3 g1(DIM / 256, (BATCH * SEQ) / 16, 3);         // 4 x 512 x 3, 4 waves/block
  qkv_gemm<<<g1, 128, 0, stream>>>(xh, Wh, Qh, Kh, Vth);

  const size_t shmem = (size_t)16 * SEQ * 4 + (size_t)16 * SEQ * 2;  // 192 KB (WGP has 320 KB)
  (void)hipFuncSetAttribute((const void*)attention,
                            hipFuncAttributeMaxDynamicSharedMemorySize, (int)shmem);
  dim3 g2(SEQ / 16, BATCH);                          // 128 x 4 blocks, 8 waves each
  attention<<<g2, 256, shmem, stream>>>(Qh, Kh, Vth, (float*)d_out);
}
